// GSJKN_6141803233697
// MI455X (gfx1250) — compile-verified, matrix operand order checked by reading, not run on hardware
//
#include <hip/hip_runtime.h>
#include <hip/hip_bf16.h>
#include <math.h>

#define NNODES   131072
#define NEDGES   1048576
#define HDIM     64
#define NLAYERS  3
#define RDIM     128
#define JKDIM    192
#define TSTEPS   512
#define NGRAPHS  256
#define GDIM     384   /* 3*R */
#define NF       (NEDGES + NNODES)   /* edges + self loops */

typedef __attribute__((ext_vector_type(16))) _Float16 v16h;
typedef __attribute__((ext_vector_type(8)))  _Float16 v8h;
typedef __attribute__((ext_vector_type(8)))  float    v8f;
typedef __attribute__((ext_vector_type(4)))  unsigned u32x4;
typedef __attribute__((ext_vector_type(8)))  int      i32x8;
typedef __attribute__((ext_vector_type(4)))  int      i32x4;

#define CDIV(a,b) (((a)+(b)-1)/(b))

// Tensor Data Mover: use the clang builtin when present. Arity differs across
// toolchains (ROCm 7.2 clang-22: 5 args; amdgpu-toolchain clang-23: 6 args).
#if defined(__HIP_DEVICE_COMPILE__) && defined(__has_builtin)
# if __has_builtin(__builtin_amdgcn_tensor_load_to_lds) && \
     __has_builtin(__builtin_amdgcn_s_wait_tensorcnt)
#  define USE_TDM 1
# endif
#endif
#ifndef USE_TDM
# define USE_TDM 0
#endif

// ---------------------------------------------------------------- utilities
__global__ void k_fill_u32(unsigned* __restrict__ p, unsigned v, int n) {
  int i = blockIdx.x * blockDim.x + threadIdx.x;
  if (i < n) p[i] = v;
}

__global__ void k_f32_to_f16(const float* __restrict__ src, _Float16* __restrict__ dst, int n) {
  int i = blockIdx.x * blockDim.x + threadIdx.x;
  if (i < n) dst[i] = (_Float16)src[i];
}

// Stage a weight matrix as column-major B (Bc[n*Kp + k] = B[k][n]) in f16 with
// row stride Kp = K+8 (16B pad -> LDS bank rotation; TDM copies the pad too).
// wTransposed==1: W is [Ncols, K] (PyTorch wih); else W is [K, Ncols].
__global__ void k_stage_B(const float* __restrict__ W, _Float16* __restrict__ Bc,
                          int K, int Ncols, int wTransposed) {
  int Kp = K + 8;
  int i = blockIdx.x * blockDim.x + threadIdx.x;
  if (i >= Ncols * Kp) return;
  int n = i / Kp, k = i % Kp;
  float v = 0.f;
  if (k < K) v = wTransposed ? W[(size_t)n * K + k] : W[(size_t)k * Ncols + n];
  Bc[i] = (_Float16)v;
}

// ------------------------------------------------------------- embeddings
__global__ void k_embed(const float* __restrict__ xin, const float* __restrict__ W,
                        const float* __restrict__ b, float* __restrict__ out, int rows) {
  int i = blockIdx.x * blockDim.x + threadIdx.x;
  if (i >= rows * HDIM) return;
  int r = i >> 6, c = i & 63;
  out[i] = xin[r * 2 + 0] * W[c] + xin[r * 2 + 1] * W[64 + c] + b[c];
}

// ----------------------------------------------------- self-loop features
__global__ void k_deg(const int* __restrict__ ei, float* __restrict__ deg) {
  int f = blockIdx.x * blockDim.x + threadIdx.x;
  if (f < NEDGES) atomicAdd(&deg[ei[NEDGES + f]], 1.0f);
}

__global__ void k_eloop_sum(const int* __restrict__ ei, const float* __restrict__ e,
                            float* __restrict__ el) {
  int i = blockIdx.x * blockDim.x + threadIdx.x;
  if (i >= NEDGES * HDIM) return;
  int f = i >> 6, c = i & 63;
  atomicAdd(&el[(size_t)ei[NEDGES + f] * HDIM + c], e[i]);
}

__global__ void k_eloop_div(const float* __restrict__ deg, float* __restrict__ el) {
  int i = blockIdx.x * blockDim.x + threadIdx.x;
  if (i >= NNODES * HDIM) return;
  el[i] /= fmaxf(deg[i >> 6], 1.0f);
}

// ------------------------------------------------------------ WMMA GEMM
// C[M, Ncols] = act(A[M,K] @ B[K,Ncols] + bias)
// block = 256 threads (8 waves); wave w -> rows [row0+16w,+16), cols [col0,+16)
// B strip lives column-major in LDS (padded stride); fragments are b128 loads.
template <int K, int ACT, int HAS_BIAS>
__global__ void k_wmma_gemm(const _Float16* __restrict__ A,
                            const _Float16* __restrict__ Bc,   // [Ncols][Kp]
                            const float* __restrict__ bias,
                            float* __restrict__ C, int M, int Ncols) {
  constexpr int Kp = K + 8;
  __shared__ _Float16 sB[16 * Kp];
  const int col0 = blockIdx.x * 16;
  const int row0 = blockIdx.y * 128;

#if USE_TDM
  if (threadIdx.x < 32) {
    // ---- Tensor DMA: 2-D tile, 16 rows x Kp halves, into LDS.
    unsigned ldsoff = (unsigned)(size_t)&sB[0];
    unsigned long long ga = (unsigned long long)(size_t)(Bc + (size_t)col0 * Kp);
    u32x4 g0;
    g0[0] = 1u;                                     // count=1, user mode
    g0[1] = ldsoff;                                 // lds_addr (bytes)
    g0[2] = (unsigned)ga;                           // global_addr lo
    g0[3] = (unsigned)(ga >> 32) | (2u << 30);      // global_addr hi | type=2
    i32x8 g1;
    g1[0] = (int)(1u << 16);                        // data_size = 2 bytes
    g1[1] = (int)((unsigned)Kp << 16);              // tensor_dim0[15:0]
    g1[2] = (int)((unsigned)Ncols << 16);           // td0 hi(0) | tensor_dim1 lo
    g1[3] = (int)((unsigned)Kp << 16);              // td1 hi(0) | tile_dim0 = Kp
    g1[4] = 16;                                     // tile_dim1 = 16 rows
    g1[5] = Kp;                                     // tensor_dim0_stride lo32
    g1[6] = 0;
    g1[7] = 0;
    i32x4 gz = {0, 0, 0, 0};
#if defined(__clang_major__) && (__clang_major__ >= 23)
    i32x8 gz8 = {0, 0, 0, 0, 0, 0, 0, 0};
    __builtin_amdgcn_tensor_load_to_lds(g0, g1, gz, gz, gz8, 0);
#else
    __builtin_amdgcn_tensor_load_to_lds(g0, g1, gz, gz, 0);
#endif
    __builtin_amdgcn_s_wait_tensorcnt(0);
  }
#else
  {
    int nrow = threadIdx.x >> 4;                    // 16 threads per B row
    int l16  = threadIdx.x & 15;
    for (int k = l16 * 8; k < K; k += 128)
      *(v8h*)(sB + nrow * Kp + k) = *(const v8h*)(Bc + (size_t)(col0 + nrow) * Kp + k);
  }
#endif
  __syncthreads();

  const int wave = threadIdx.x >> 5;
  const int lane = threadIdx.x & 31;
  const int r    = lane & 15;                       // A row / B col / D col
  const int kb   = (lane < 16) ? 0 : 8;             // per-lane K base (ISA layout)
  const _Float16* __restrict__ Arow = A + (size_t)(row0 + wave * 16 + r) * K;
  const _Float16* __restrict__ Brow = sB + r * Kp;

  v8f acc = {};
#pragma unroll
  for (int k0 = 0; k0 < K; k0 += 32) {
    v8h a0 = *(const v8h*)(Arow + k0 + kb);         // elems 0..7 : k = kb..kb+7
    v8h a1 = *(const v8h*)(Arow + k0 + kb + 16);    // elems 8..15: k = kb+16..kb+23
    v8h b0 = *(const v8h*)(Brow + k0 + kb);
    v8h b1 = *(const v8h*)(Brow + k0 + kb + 16);
    v16h a, b;
#pragma unroll
    for (int i = 0; i < 8; ++i) {
      a[i] = a0[i]; a[i + 8] = a1[i];
      b[i] = b0[i]; b[i + 8] = b1[i];
    }
    acc = __builtin_amdgcn_wmma_f32_16x16x32_f16(false, a, false, b,
                                                 (short)0, acc, false, false);
  }

  float bv = 0.f;
  if (HAS_BIAS) bv = bias[col0 + r];
#pragma unroll
  for (int j = 0; j < 8; ++j) {
    int m = row0 + wave * 16 + j + ((lane < 16) ? 0 : 8);
    float v = acc[j];
    if (HAS_BIAS) v += bv;
    if (ACT == 1) v = (v > 0.f) ? v : 0.01f * v;    // leaky_relu 0.01
    if (ACT == 2) v = fmaxf(v, 0.f);                // relu
    C[(size_t)m * Ncols + col0 + r] = v;
  }
}

// ------------------------------------------------------------ GAT pieces
__global__ void k_vmake(const float* __restrict__ Wedge, const float* __restrict__ att,
                        float* __restrict__ v) {
  int k = threadIdx.x;                              // 64 threads
  float s = 0.f;
  for (int j = 0; j < 64; ++j) s += Wedge[k * 64 + j] * att[j];
  v[k] = s;
}

__global__ void k_att_nodes(const float* __restrict__ xp, const float* __restrict__ att_s,
                            const float* __restrict__ att_d, float* __restrict__ a_s,
                            float* __restrict__ a_d) {
  int n = blockIdx.x * blockDim.x + threadIdx.x;
  if (n >= NNODES) return;
  const float* row = xp + (size_t)n * HDIM;
  float s = 0.f, d = 0.f;
  for (int c = 0; c < HDIM; ++c) { s += row[c] * att_s[c]; d += row[c] * att_d[c]; }
  a_s[n] = s; a_d[n] = d;
}

__device__ __forceinline__ unsigned f2ord(float f) {
  unsigned u = __float_as_uint(f);
  return (u & 0x80000000u) ? ~u : (u | 0x80000000u);
}
__device__ __forceinline__ float ord2f(unsigned u) {
  return (u & 0x80000000u) ? __uint_as_float(u & 0x7fffffffu) : __uint_as_float(~u);
}

__global__ void k_alpha(const int* __restrict__ ei, const float* __restrict__ e,
                        const float* __restrict__ el, const float* __restrict__ v,
                        const float* __restrict__ a_s, const float* __restrict__ a_d,
                        float* __restrict__ alpha, unsigned* __restrict__ m_ord) {
  int f = blockIdx.x * blockDim.x + threadIdx.x;
  if (f >= NF) return;
  int s, d; const float* erow;
  if (f < NEDGES) { s = ei[f]; d = ei[NEDGES + f]; erow = e + (size_t)f * HDIM; }
  else            { s = d = f - NEDGES;           erow = el + (size_t)(f - NEDGES) * HDIM; }
  float ae = 0.f;
  for (int c = 0; c < HDIM; ++c) ae += erow[c] * v[c];
  float al = a_s[s] + a_d[d] + ae;
  al = (al > 0.f) ? al : 0.2f * al;                 // leaky_relu 0.2
  alpha[f] = al;
  atomicMax(&m_ord[d], f2ord(al));                  // segment max via ordered-uint
}

__global__ void k_expsum(const int* __restrict__ ei, const float* __restrict__ alpha,
                         const unsigned* __restrict__ m_ord, float* __restrict__ exv,
                         float* __restrict__ den) {
  int f = blockIdx.x * blockDim.x + threadIdx.x;
  if (f >= NF) return;
  int d = (f < NEDGES) ? ei[NEDGES + f] : f - NEDGES;
  float x = __expf(alpha[f] - ord2f(m_ord[d]));
  exv[f] = x;
  atomicAdd(&den[d], x);
}

__global__ void k_agg(const int* __restrict__ ei, const float* __restrict__ exv,
                      const float* __restrict__ den, const float* __restrict__ xp,
                      float* __restrict__ agg) {
  int i = blockIdx.x * blockDim.x + threadIdx.x;
  if (i >= NF * HDIM) return;
  int f = i >> 6, c = i & 63;
  int s, d;
  if (f < NEDGES) { s = ei[f]; d = ei[NEDGES + f]; }
  else            { s = d = f - NEDGES; }
  float w = exv[f] / den[d];
  atomicAdd(&agg[(size_t)d * HDIM + c], w * xp[(size_t)s * HDIM + c]);
}

__global__ void k_gat_final(const float* __restrict__ agg, const float* __restrict__ bias,
                            float* __restrict__ h, float* __restrict__ jk, int l) {
  int i = blockIdx.x * blockDim.x + threadIdx.x;
  if (i >= NNODES * HDIM) return;
  int n = i >> 6, c = i & 63;
  float v = agg[i] + bias[c];
  v = (v > 0.f) ? v : 0.01f * v;
  h[i] = v;
  jk[(size_t)n * JKDIM + l * HDIM + c] = v;
}

// ------------------------------------------------------------ GRU scan
__global__ void k_gru_scan(const float* __restrict__ gx,   // [N, 384], n = b*512 + t
                           const float* __restrict__ whh,  // [384, 128]
                           const float* __restrict__ bhh,  // [384]
                           float* __restrict__ outcat,     // [N, 256]
                           int reverse, int out_off) {
  __shared__ float sh[RDIM];
  const int b = blockIdx.x;
  const int j = threadIdx.x;                        // 128 threads
  sh[j] = 0.f;
  __syncthreads();
  const float br = bhh[j], bz = bhh[RDIM + j], bn = bhh[2 * RDIM + j];
  const float* wr = whh + (size_t)j * RDIM;
  const float* wz = whh + (size_t)(RDIM + j) * RDIM;
  const float* wn = whh + (size_t)(2 * RDIM + j) * RDIM;

  for (int step = 0; step < TSTEPS; ++step) {
    int t = reverse ? (TSTEPS - 1 - step) : step;
    size_t n = (size_t)b * TSTEPS + t;
    if (step + 1 < TSTEPS) {
      int tn = reverse ? (t - 1) : (t + 1);
      __builtin_prefetch(gx + ((size_t)b * TSTEPS + tn) * GDIM + j, 0, 3);
    }
    float ghr = br, ghz = bz, ghn = bn;
#pragma unroll 8
    for (int k = 0; k < RDIM; ++k) {
      float hk = sh[k];
      ghr += wr[k] * hk;
      ghz += wz[k] * hk;
      ghn += wn[k] * hk;
    }
    const float* g = gx + n * GDIM;
    float rr = 1.f / (1.f + __expf(-(g[j] + ghr)));
    float zz = 1.f / (1.f + __expf(-(g[RDIM + j] + ghz)));
    float nn = tanhf(g[2 * RDIM + j] + rr * ghn);
    float hn = (1.f - zz) * nn + zz * sh[j];
    __syncthreads();
    sh[j] = hn;
    __syncthreads();
    outcat[n * 256 + out_off + j] = hn;
  }
}

// ------------------------------------------------------------ head tail
__global__ void k_head_out(const float* __restrict__ hid, const float* __restrict__ W2,
                           const float* __restrict__ b2, float* __restrict__ out) {
  int n = blockIdx.x * blockDim.x + threadIdx.x;
  if (n >= NNODES) return;
  const float* row = hid + (size_t)n * 64;
  float o0 = b2[0], o1 = b2[1];
  for (int c = 0; c < 64; ++c) { o0 += row[c] * W2[c * 2]; o1 += row[c] * W2[c * 2 + 1]; }
  out[n * 2 + 0] = o0;
  out[n * 2 + 1] = o1;
}

// ---------------------------------------------------------------- driver
extern "C" void kernel_launch(void* const* d_in, const int* in_sizes, int n_in,
                              void* d_out, int out_size, void* d_ws, size_t ws_size,
                              hipStream_t stream) {
  (void)in_sizes; (void)n_in; (void)out_size; (void)ws_size;

  const float* x          = (const float*)d_in[0];
  const float* edge_attr  = (const float*)d_in[1];
  const float* node_W     = (const float*)d_in[2];
  const float* node_b     = (const float*)d_in[3];
  const float* edgeemb_W  = (const float*)d_in[4];
  const float* edgeemb_b  = (const float*)d_in[5];
  const float* gat_W      = (const float*)d_in[6];
  const float* gat_Wedge  = (const float*)d_in[7];
  const float* att_src    = (const float*)d_in[8];
  const float* att_dst    = (const float*)d_in[9];
  const float* att_edge   = (const float*)d_in[10];
  const float* gat_b      = (const float*)d_in[11];
  const float* wih_f      = (const float*)d_in[12];
  const float* whh_f      = (const float*)d_in[13];
  const float* bih_f      = (const float*)d_in[14];
  const float* bhh_f      = (const float*)d_in[15];
  const float* wih_b      = (const float*)d_in[16];
  const float* whh_b      = (const float*)d_in[17];
  const float* bih_b      = (const float*)d_in[18];
  const float* bhh_b      = (const float*)d_in[19];
  const float* head_W1    = (const float*)d_in[20];
  const float* head_b1    = (const float*)d_in[21];
  const float* head_W2    = (const float*)d_in[22];
  const float* head_b2    = (const float*)d_in[23];
  const int*   ei         = (const int*)d_in[24];

  char* wsb = (char*)d_ws;
  size_t off = 0;
  auto carve = [&](size_t bytes) -> void* {
    void* p = wsb + off;
    off += (bytes + 255) & ~(size_t)255;
    return p;
  };

  float*    h      = (float*)carve((size_t)NNODES * HDIM * 4);
  float*    e      = (float*)carve((size_t)NEDGES * HDIM * 4);
  float*    deg    = (float*)carve((size_t)NNODES * 4);
  float*    e_loop = (float*)carve((size_t)NNODES * HDIM * 4);
  float*    xp     = (float*)carve((size_t)NNODES * HDIM * 4);
  float*    a_s    = (float*)carve((size_t)NNODES * 4);
  float*    a_d    = (float*)carve((size_t)NNODES * 4);
  float*    vvec   = (float*)carve(64 * 4);
  float*    alpha  = (float*)carve((size_t)NF * 4);
  float*    exv    = (float*)carve((size_t)NF * 4);
  unsigned* m_ord  = (unsigned*)carve((size_t)NNODES * 4);
  float*    den    = (float*)carve((size_t)NNODES * 4);
  float*    agg    = (float*)carve((size_t)NNODES * HDIM * 4);
  float*    jk     = (float*)carve((size_t)NNODES * JKDIM * 4);
  _Float16* Ah     = (_Float16*)carve((size_t)NNODES * 256 * 2);
  _Float16* Bc     = (_Float16*)carve((size_t)GDIM * 264 * 2 + 256); // max Ncols*Kp
  float*    gx     = (float*)carve((size_t)NNODES * GDIM * 4);
  float*    rnncat = (float*)carve((size_t)NNODES * 256 * 4);
  float*    hid    = (float*)carve((size_t)NNODES * 64 * 4);

  const int T = 256;

  // ---- embeddings + self-loop edge features
  k_fill_u32<<<CDIV(NNODES, T), T, 0, stream>>>((unsigned*)deg, 0u, NNODES);
  k_fill_u32<<<CDIV(NNODES * HDIM, T), T, 0, stream>>>((unsigned*)e_loop, 0u, NNODES * HDIM);
  k_embed<<<CDIV(NNODES * HDIM, T), T, 0, stream>>>(x, node_W, node_b, h, NNODES);
  k_embed<<<CDIV(NEDGES * HDIM, T), T, 0, stream>>>(edge_attr, edgeemb_W, edgeemb_b, e, NEDGES);
  k_deg<<<CDIV(NEDGES, T), T, 0, stream>>>(ei, deg);
  k_eloop_sum<<<CDIV(NEDGES * HDIM, T), T, 0, stream>>>(ei, e, e_loop);
  k_eloop_div<<<CDIV(NNODES * HDIM, T), T, 0, stream>>>(deg, e_loop);

  // ---- GAT layers
  for (int l = 0; l < NLAYERS; ++l) {
    k_f32_to_f16<<<CDIV(NNODES * HDIM, T), T, 0, stream>>>(h, Ah, NNODES * HDIM);
    k_stage_B<<<CDIV(HDIM * (HDIM + 8), T), T, 0, stream>>>(
        gat_W + (size_t)l * HDIM * HDIM, Bc, HDIM, HDIM, 0);
    dim3 g1(HDIM / 16, NNODES / 128);
    k_wmma_gemm<64, 0, 0><<<g1, 256, 0, stream>>>(Ah, Bc, nullptr, xp, NNODES, HDIM);

    k_att_nodes<<<CDIV(NNODES, T), T, 0, stream>>>(xp, att_src + l * HDIM, att_dst + l * HDIM, a_s, a_d);
    k_vmake<<<1, 64, 0, stream>>>(gat_Wedge + (size_t)l * HDIM * HDIM, att_edge + l * HDIM, vvec);

    k_fill_u32<<<CDIV(NNODES, T), T, 0, stream>>>(m_ord, 0u, NNODES);
    k_fill_u32<<<CDIV(NNODES, T), T, 0, stream>>>((unsigned*)den, 0u, NNODES);
    k_fill_u32<<<CDIV(NNODES * HDIM, T), T, 0, stream>>>((unsigned*)agg, 0u, NNODES * HDIM);

    k_alpha<<<CDIV(NF, T), T, 0, stream>>>(ei, e, e_loop, vvec, a_s, a_d, alpha, m_ord);
    k_expsum<<<CDIV(NF, T), T, 0, stream>>>(ei, alpha, m_ord, exv, den);
    k_agg<<<CDIV(NF * HDIM, T), T, 0, stream>>>(ei, exv, den, xp, agg);
    k_gat_final<<<CDIV(NNODES * HDIM, T), T, 0, stream>>>(agg, gat_b + l * HDIM, h, jk, l);
  }

  // ---- bidirectional GRU
  k_f32_to_f16<<<CDIV(NNODES * JKDIM, T), T, 0, stream>>>(jk, Ah, NNODES * JKDIM);
  dim3 g2(GDIM / 16, NNODES / 128);

  k_stage_B<<<CDIV(GDIM * (JKDIM + 8), T), T, 0, stream>>>(wih_f, Bc, JKDIM, GDIM, 1);
  k_wmma_gemm<192, 0, 1><<<g2, 256, 0, stream>>>(Ah, Bc, bih_f, gx, NNODES, GDIM);
  k_gru_scan<<<NGRAPHS, RDIM, 0, stream>>>(gx, whh_f, bhh_f, rnncat, 0, 0);

  k_stage_B<<<CDIV(GDIM * (JKDIM + 8), T), T, 0, stream>>>(wih_b, Bc, JKDIM, GDIM, 1);
  k_wmma_gemm<192, 0, 1><<<g2, 256, 0, stream>>>(Ah, Bc, bih_b, gx, NNODES, GDIM);
  k_gru_scan<<<NGRAPHS, RDIM, 0, stream>>>(gx, whh_b, bhh_b, rnncat, 1, RDIM);

  // ---- head
  k_f32_to_f16<<<CDIV(NNODES * 256, T), T, 0, stream>>>(rnncat, Ah, NNODES * 256);
  k_stage_B<<<CDIV(64 * (256 + 8), T), T, 0, stream>>>(head_W1, Bc, 256, 64, 0);
  dim3 g3(64 / 16, NNODES / 128);
  k_wmma_gemm<256, 2, 1><<<g3, 256, 0, stream>>>(Ah, Bc, head_b1, hid, NNODES, 64);
  k_head_out<<<CDIV(NNODES, T), T, 0, stream>>>(hid, head_W2, head_b2, (float*)d_out);
}
